// HyperComputeModule_80771154968988
// MI455X (gfx1250) — compile-verified, matrix operand order checked by reading, not run on hardware
//
#include <hip/hip_runtime.h>
#include <hip/hip_bf16.h>
#include <math.h>

// Problem constants (from reference): B=4, C=C2=128, H=W=64, N=4096
#define BATCH 4
#define CH    128
#define NPTS  4096
#define THR2  256.0f   // THRESHOLD^2 ; dist<16  <=>  d2<256 (incl. clamped negatives)
#define BNEPS 1e-5f

typedef _Float16 half_t;
typedef __attribute__((ext_vector_type(16))) _Float16 v16h;
typedef __attribute__((ext_vector_type(8)))  _Float16 v8h;
typedef __attribute__((ext_vector_type(8)))  float    v8f;

#define DEVINL __device__ __forceinline__

// WMMA f32 = f16 x f16 + f32, 16x16x32 (codegen-confirmed builtin)
DEVINL v8f wmma_f16(v16h a, v16h b, v8f c) {
  return __builtin_amdgcn_wmma_f32_16x16x32_f16(false, a, false, b, (short)0, c,
                                                false, false);
}

// Per-lane A/B fragment: 8 halves at p, 8 halves at p+16 (ISA 16-bit 16x32 layout)
DEVINL v16h load_ab(const half_t* p) {
  v8h lo = *(const v8h*)(p);
  v8h hi = *(const v8h*)(p + 16);
  return __builtin_shufflevector(lo, hi, 0, 1, 2, 3, 4, 5, 6, 7,
                                 8, 9, 10, 11, 12, 13, 14, 15);
}

// Expand 2x8 incidence bits -> v16h of 0/1 halves (A fragment)
DEVINL v16h expand16(unsigned wb, int hi) {
  unsigned b0 = (wb >> (hi * 8)) & 0xFFu;
  unsigned b1 = (wb >> (hi * 8 + 16)) & 0xFFu;
  v16h r;
#pragma unroll
  for (int t = 0; t < 8; ++t) {
    r[t]     = (half_t)(int)((b0 >> t) & 1u);
    r[t + 8] = (half_t)(int)((b1 >> t) & 1u);
  }
  return r;
}

// ---------------------------------------------------------------- small preps
__global__ void __launch_bounds__(256) k_convert_w(const float* __restrict__ W,
                                                   half_t* __restrict__ w16) {
  int i = blockIdx.x * 256 + threadIdx.x;
  if (i < CH * CH) w16[i] = (half_t)W[i];
}

// xf16[b][n][c] = (half) x[b][c][n]  (tiled transpose through LDS)
__global__ void __launch_bounds__(256) k_transpose(const float* __restrict__ x,
                                                   half_t* __restrict__ xf16) {
  __shared__ float tile[32][33];
  const int n0 = blockIdx.x * 32, c0 = blockIdx.y * 32, b = blockIdx.z;
  const int tx = threadIdx.x, ty = threadIdx.y;
#pragma unroll
  for (int q = 0; q < 4; ++q)
    tile[ty + 8 * q][tx] =
        x[((size_t)(b * CH + c0 + ty + 8 * q)) * NPTS + n0 + tx];
  __syncthreads();
#pragma unroll
  for (int q = 0; q < 4; ++q)
    xf16[((size_t)(b * NPTS + n0 + ty + 8 * q)) * CH + c0 + tx] =
        (half_t)tile[tx][ty + 8 * q];
}

// sq[b][n] = sum_c x[b][c][n]^2   (coalesced along n)
__global__ void __launch_bounds__(256) k_sq(const float* __restrict__ x,
                                            float* __restrict__ sq) {
  const int b = blockIdx.y;
  const int n = blockIdx.x * 256 + threadIdx.x;
  float s = 0.f;
  for (int c = 0; c < CH; ++c) {
    float v = x[((size_t)(b * CH + c)) * NPTS + n];
    s += v * v;
  }
  sq[b * NPTS + n] = s;
}

// -------------------------------------------------- Gram + incidence bitmask
// One wave = strip of 4 16x16 tiles (A fragment kept in regs).
// inc16[b][i][j/16] : 16 incidence bits per ushort.
__global__ void __launch_bounds__(256) k_gram(const half_t* __restrict__ xf16,
                                              const float* __restrict__ sq,
                                              unsigned short* __restrict__ inc16) {
  const int b = blockIdx.z;
  const int i0 = blockIdx.y * 16;
  const int wave = threadIdx.x >> 5;
  const int lane = threadIdx.x & 31;
  const int r = lane & 15, hi = lane >> 4;

  const half_t* arow = xf16 + ((size_t)(b * NPTS + i0 + r)) * CH + hi * 8;
  v16h A[4];
#pragma unroll
  for (int kk = 0; kk < 4; ++kk) A[kk] = load_ab(arow + kk * 32);
  const v8f sqi = *(const v8f*)(sq + b * NPTS + i0 + hi * 8);

#pragma unroll
  for (int t = 0; t < 4; ++t) {
    const int jt = blockIdx.x * 32 + wave * 4 + t;  // col tile 0..255
    const int j0 = jt * 16;
    const half_t* brow = xf16 + ((size_t)(b * NPTS + j0 + r)) * CH + hi * 8;
    v8f acc = {0.f, 0.f, 0.f, 0.f, 0.f, 0.f, 0.f, 0.f};
#pragma unroll
    for (int kk = 0; kk < 4; ++kk)
      acc = wmma_f16(A[kk], load_ab(brow + kk * 32), acc);

    const float sqj = sq[b * NPTS + j0 + r];
    unsigned sel = 0;
#pragma unroll
    for (int v = 0; v < 8; ++v) {
      float d2 = sqi[v] + sqj - 2.0f * acc[v];
      unsigned bl = __builtin_amdgcn_ballot_w32(d2 < THR2);
      // D layout: lanes0-15 row v, lanes16-31 row v+8 (cols = lane)
      if (lane == v) sel = bl & 0xFFFFu;
      if (lane == v + 8) sel = bl >> 16;
    }
    if (lane < 16)
      inc16[((size_t)(b * NPTS + i0 + lane)) * (NPTS / 16) + jt] =
          (unsigned short)sel;
  }
}

// deg[row] = popcount of incidence row (symmetric => deg_e == deg_v)
__global__ void __launch_bounds__(128) k_deg(const unsigned* __restrict__ incW,
                                             float* __restrict__ deg) {
  __shared__ int red[128];
  const int row = blockIdx.x, t = threadIdx.x;
  red[t] = __popc(incW[(size_t)row * (NPTS / 32) + t]);
  __syncthreads();
  for (int s = 64; s > 0; s >>= 1) {
    if (t < s) red[t] += red[t + s];
    __syncthreads();
  }
  if (t == 0) deg[row] = (float)red[0];
}

// ----------------------------------------------- xe = xf @ W^T + bias (WMMA)
// Output transposed: xeT[b][d][n]  (contiguous v8h epilogue stores)
__global__ void __launch_bounds__(256) k_xe(const half_t* __restrict__ xf16,
                                            const half_t* __restrict__ w16,
                                            const float* __restrict__ bias,
                                            half_t* __restrict__ xeT) {
  const int row0 = blockIdx.x * 16;              // global row over B*N
  const int d0 = (threadIdx.x >> 5) * 16;        // 8 waves -> 8 d-tiles
  const int lane = threadIdx.x & 31;
  const int r = lane & 15, hi = lane >> 4;

  const half_t* arow = xf16 + ((size_t)(row0 + r)) * CH + hi * 8;
  const half_t* brow = w16 + ((size_t)(d0 + r)) * CH + hi * 8;  // B^T = W
  v8f acc = {0.f, 0.f, 0.f, 0.f, 0.f, 0.f, 0.f, 0.f};
#pragma unroll
  for (int kk = 0; kk < 4; ++kk)
    acc = wmma_f16(load_ab(arow + kk * 32), load_ab(brow + kk * 32), acc);

  const float bs = bias[d0 + r];
  v8h o;
#pragma unroll
  for (int v = 0; v < 8; ++v) o[v] = (half_t)(acc[v] + bs);
  const int batch = row0 >> 12, nn = row0 & (NPTS - 1);
  *(v8h*)(xeT + ((size_t)(batch * CH + d0 + r)) * NPTS + nn + hi * 8) = o;
}

// ------------------------- edge = diag(1/deg) * inc @ xe   (bitmask A, WMMA)
// One wave = 16x128 output strip: expand incidence bits ONCE per K-step,
// feed 8 independent WMMAs (all 8 c-tiles) -> XDL-bound, not VALU-bound.
__global__ void __launch_bounds__(256) k_edge(const unsigned* __restrict__ incW,
                                              const half_t* __restrict__ xeT,
                                              const float* __restrict__ deg,
                                              half_t* __restrict__ edgeT) {
  const int b = blockIdx.y;
  const int e0 = (blockIdx.x * 8 + (threadIdx.x >> 5)) * 16;
  const int lane = threadIdx.x & 31;
  const int r = lane & 15, hi = lane >> 4;

  const unsigned* wrow = incW + ((size_t)(b * NPTS + e0 + r)) * (NPTS / 32);
  const half_t* bbase = xeT + ((size_t)(b * CH + r)) * NPTS + hi * 8;
  v8f acc[8];
#pragma unroll
  for (int cc = 0; cc < 8; ++cc)
    acc[cc] = (v8f){0.f, 0.f, 0.f, 0.f, 0.f, 0.f, 0.f, 0.f};

  for (int kk = 0; kk < NPTS / 32; ++kk) {
    const v16h A = expand16(wrow[kk], hi);
    const half_t* bp = bbase + kk * 32;
#pragma unroll
    for (int cc = 0; cc < 8; ++cc)
      acc[cc] = wmma_f16(A, load_ab(bp + (size_t)cc * 16 * NPTS), acc[cc]);
  }

  const v8f dg = *(const v8f*)(deg + b * NPTS + e0 + hi * 8);
  v8f inv;
#pragma unroll
  for (int v = 0; v < 8; ++v) inv[v] = (dg[v] > 0.f) ? (1.0f / dg[v]) : 0.f;
#pragma unroll
  for (int cc = 0; cc < 8; ++cc) {
    v8h o;
#pragma unroll
    for (int v = 0; v < 8; ++v) o[v] = (half_t)(acc[cc][v] * inv[v]);
    *(v8h*)(edgeT + ((size_t)(b * CH + cc * 16 + r)) * NPTS + e0 + hi * 8) = o;
  }
}

// ------- out[b][c][n] = (1/deg_v) * (inc @ edge) + x[b][c][n]   (f32, WMMA)
__global__ void __launch_bounds__(256) k_vert(const unsigned* __restrict__ incW,
                                              const half_t* __restrict__ edgeT,
                                              const float* __restrict__ deg,
                                              const float* __restrict__ x,
                                              float* __restrict__ out32) {
  const int b = blockIdx.y;
  const int n0 = (blockIdx.x * 8 + (threadIdx.x >> 5)) * 16;
  const int lane = threadIdx.x & 31;
  const int r = lane & 15, hi = lane >> 4;

  const unsigned* wrow = incW + ((size_t)(b * NPTS + n0 + r)) * (NPTS / 32);
  const half_t* bbase = edgeT + ((size_t)(b * CH + r)) * NPTS + hi * 8;
  v8f acc[8];
#pragma unroll
  for (int cc = 0; cc < 8; ++cc)
    acc[cc] = (v8f){0.f, 0.f, 0.f, 0.f, 0.f, 0.f, 0.f, 0.f};

  for (int kk = 0; kk < NPTS / 32; ++kk) {
    const v16h A = expand16(wrow[kk], hi);
    const half_t* bp = bbase + kk * 32;
#pragma unroll
    for (int cc = 0; cc < 8; ++cc)
      acc[cc] = wmma_f16(A, load_ab(bp + (size_t)cc * 16 * NPTS), acc[cc]);
  }

  const v8f dg = *(const v8f*)(deg + b * NPTS + n0 + hi * 8);
  v8f inv;
#pragma unroll
  for (int v = 0; v < 8; ++v) inv[v] = (dg[v] > 0.f) ? (1.0f / dg[v]) : 0.f;
#pragma unroll
  for (int cc = 0; cc < 8; ++cc) {
    const size_t off = ((size_t)(b * CH + cc * 16 + r)) * NPTS + n0 + hi * 8;
    const v8f xv = *(const v8f*)(x + off);
    v8f o;
#pragma unroll
    for (int v = 0; v < 8; ++v) o[v] = acc[cc][v] * inv[v] + xv[v];
    *(v8f*)(out32 + off) = o;
  }
}

// --------------------------------------------------------------- BatchNorm
__global__ void __launch_bounds__(256) k_bn_stats(const float* __restrict__ out32,
                                                  float* __restrict__ bnM,
                                                  float* __restrict__ bnR) {
  __shared__ float s1[256], s2[256];
  const int c = blockIdx.x, t = threadIdx.x;
  float a = 0.f, q = 0.f;
  for (int b = 0; b < BATCH; ++b) {
    const float* p = out32 + ((size_t)(b * CH + c)) * NPTS;
    for (int n = t; n < NPTS; n += 256) {
      float v = p[n];
      a += v;
      q += v * v;
    }
  }
  s1[t] = a;
  s2[t] = q;
  __syncthreads();
  for (int s = 128; s > 0; s >>= 1) {
    if (t < s) { s1[t] += s1[t + s]; s2[t] += s2[t + s]; }
    __syncthreads();
  }
  if (t == 0) {
    const float inv = 1.0f / (float)(BATCH * NPTS);
    float m = s1[0] * inv;
    float var = s2[0] * inv - m * m;
    bnM[c] = m;
    bnR[c] = rsqrtf(fmaxf(var, 0.f) + BNEPS);
  }
}

__global__ void __launch_bounds__(256) k_bn_silu(const float* __restrict__ out32,
                                                 const float* __restrict__ gamma,
                                                 const float* __restrict__ beta,
                                                 const float* __restrict__ bnM,
                                                 const float* __restrict__ bnR,
                                                 float* __restrict__ y) {
  const int idx = blockIdx.x * 256 + threadIdx.x;  // over B*C*N = 2M
  const int c = (idx >> 12) & (CH - 1);
  float v = out32[idx];
  float t = gamma[c] * ((v - bnM[c]) * bnR[c]) + beta[c];
  y[idx] = t / (1.0f + expf(-t));  // SiLU
}

// ---------------------------------------------------------------- launcher
extern "C" void kernel_launch(void* const* d_in, const int* in_sizes, int n_in,
                              void* d_out, int out_size, void* d_ws, size_t ws_size,
                              hipStream_t stream) {
  const float* x     = (const float*)d_in[0];  // [4,128,64,64]
  const float* W_fc  = (const float*)d_in[1];  // [128,128]
  const float* b_fc  = (const float*)d_in[2];  // [128]
  const float* gamma = (const float*)d_in[3];  // [128]
  const float* beta  = (const float*)d_in[4];  // [128]
  float* out = (float*)d_out;                  // [4,128,64,64] f32

  // workspace carve-up (~29.5 MB, all offsets 256B-aligned)
  char* ws = (char*)d_ws;
  size_t o = 0;
  half_t* xf16   = (half_t*)(ws + o); o += (size_t)BATCH * NPTS * CH * 2;   // 4 MB
  half_t* w16    = (half_t*)(ws + o); o += (size_t)CH * CH * 2;             // 32 KB
  half_t* xeT    = (half_t*)(ws + o); o += (size_t)BATCH * NPTS * CH * 2;   // 4 MB
  half_t* edgeT  = (half_t*)(ws + o); o += (size_t)BATCH * NPTS * CH * 2;   // 4 MB
  float* sq      = (float*)(ws + o);  o += (size_t)BATCH * NPTS * 4;        // 64 KB
  float* deg     = (float*)(ws + o);  o += (size_t)BATCH * NPTS * 4;        // 64 KB
  unsigned short* inc16 = (unsigned short*)(ws + o);
  unsigned* incW = (unsigned*)(ws + o); o += (size_t)BATCH * NPTS * (NPTS / 8); // 8 MB
  float* out32   = (float*)(ws + o);  o += (size_t)BATCH * CH * NPTS * 4;   // 8 MB
  float* bnM     = (float*)(ws + o);  o += 512;
  float* bnR     = (float*)(ws + o);  o += 512;
  (void)in_sizes; (void)n_in; (void)out_size; (void)ws_size;

  k_convert_w<<<dim3((CH * CH + 255) / 256), 256, 0, stream>>>(W_fc, w16);
  k_transpose<<<dim3(NPTS / 32, CH / 32, BATCH), dim3(32, 8), 0, stream>>>(x, xf16);
  k_sq<<<dim3(NPTS / 256, BATCH), 256, 0, stream>>>(x, sq);
  k_gram<<<dim3(8, NPTS / 16, BATCH), 256, 0, stream>>>(xf16, sq, inc16);
  k_deg<<<dim3(BATCH * NPTS), 128, 0, stream>>>(incW, deg);
  k_xe<<<dim3(BATCH * NPTS / 16), 256, 0, stream>>>(xf16, w16, b_fc, xeT);
  k_edge<<<dim3(NPTS / 128, BATCH), 256, 0, stream>>>(incW, xeT, deg, edgeT);
  k_vert<<<dim3(NPTS / 128, BATCH), 256, 0, stream>>>(incW, edgeT, deg, x, out32);
  k_bn_stats<<<dim3(CH), 256, 0, stream>>>(out32, bnM, bnR);
  k_bn_silu<<<dim3(BATCH * CH * NPTS / 256), 256, 0, stream>>>(out32, gamma, beta,
                                                               bnM, bnR, out);
}
